// FieldConditionedEquivariantAbsorberHead_80805514707442
// MI455X (gfx1250) — matrix operand order repaired and checked by reading, not compiled, not run on hardware
//
#include <hip/hip_runtime.h>
#include <hip/hip_bf16.h>
#include <math.h>

// ---------------------------------------------------------------------------
// FieldConditionedEquivariantAbsorberHead — CDNA5 (gfx1250) WMMA implementation
//
// Shapes: B=256, NE=400, E_DIM=32, FF_DIM=64, HID=256, NCOP=224, D=480,
//         INV_IN=288, OUT=1.  Rows = B*NE = 102400.
// Strategy: factor gates-MLP layer 1 into per-e and per-b partials; precompute
// per-b mean(x^2) for the norm reduction; then one fused kernel that keeps a
// 128-row activation block in LDS and does all 4 big GEMM stages with
// v_wmma_f32_16x16x32_f16 (f16 in, f32 accumulate).
// Epilogue uses v_rcp_f32 (fast sigmoid) instead of the IEEE divide sequence.
// ---------------------------------------------------------------------------

typedef __attribute__((ext_vector_type(16))) _Float16 v16h;
typedef __attribute__((ext_vector_type(8)))  _Float16 v8h;
typedef __attribute__((ext_vector_type(8)))  float    v8f;

#define B_N     256
#define NE_N    400
#define E_DIMC  32
#define FF_DIMC 64
#define HIDC    256
#define D_TOT   480
#define NCOPC   224
#define INV_IN  288
#define ROWS    (B_N * NE_N)          // 102400
#define MT      128                   // rows per workgroup
#define EPSC    1e-8f

// Fast SiLU: x * sigmoid(x) with v_exp_f32 + v_rcp_f32 (no IEEE div sequence).
__device__ __forceinline__ float silu_f(float x) {
  return x * __builtin_amdgcn_rcpf(1.0f + __expf(-x));
}

// ---------------------------------------------------------------------------
// Prep kernels (tiny, run every launch; deterministic)
// ---------------------------------------------------------------------------

// WT[n*K + k] = (f16) W[k*N + n]   (row-major [K,N] f32 -> row-major [N,K] f16)
__global__ void k_transpose_f16(const float* __restrict__ W,
                                _Float16* __restrict__ WT, int K, int N) {
  int i = blockIdx.x * blockDim.x + threadIdx.x;
  if (i >= K * N) return;
  int n = i / K, k = i % K;
  WT[i] = (_Float16)W[k * N + n];
}

// Ae[e,n] = sum_{k<32} e_feat[e,k] * mW1[k,n]
__global__ void k_ae(const float* __restrict__ e_feat,
                     const float* __restrict__ mW1,
                     float* __restrict__ Ae) {
  int i = blockIdx.x * blockDim.x + threadIdx.x;
  if (i >= NE_N * HIDC) return;
  int e = i / HIDC, n = i % HIDC;
  float s = 0.f;
#pragma unroll 8
  for (int k = 0; k < E_DIMC; ++k)
    s += e_feat[e * E_DIMC + k] * mW1[k * HIDC + n];
  Ae[i] = s;
}

// Ab[b,n] = sum_{k<64} field[b,k] * mW1[32+k, n]
__global__ void k_ab(const float* __restrict__ field,
                     const float* __restrict__ mW1,
                     float* __restrict__ Ab) {
  int i = blockIdx.x * blockDim.x + threadIdx.x;
  if (i >= B_N * HIDC) return;
  int b = i / HIDC, n = i % HIDC;
  float s = 0.f;
#pragma unroll 8
  for (int k = 0; k < FF_DIMC; ++k)
    s += field[b * FF_DIMC + k] * mW1[(E_DIMC + k) * HIDC + n];
  Ab[i] = s;
}

// msq1[b,j] = mean_{d<3} x[b,128+3j+d]^2   (64 copies, l=1)
// msq2[b,j] = mean_{d<5} x[b,320+5j+d]^2   (32 copies, l=2)
__global__ void k_msq(const float* __restrict__ h,
                      float* __restrict__ msq1, float* __restrict__ msq2) {
  int i = blockIdx.x * blockDim.x + threadIdx.x;
  if (i >= B_N * 96) return;
  int b = i / 96, j = i % 96;
  if (j < 64) {
    float s = 0.f;
#pragma unroll
    for (int d = 0; d < 3; ++d) {
      float v = h[b * D_TOT + 128 + 3 * j + d];
      s += v * v;
    }
    msq1[b * 64 + j] = s * (1.0f / 3.0f);
  } else {
    int j2 = j - 64;
    float s = 0.f;
#pragma unroll
    for (int d = 0; d < 5; ++d) {
      float v = h[b * D_TOT + 320 + 5 * j2 + d];
      s += v * v;
    }
    msq2[b * 32 + j2] = s * (1.0f / 5.0f);
  }
}

// ---------------------------------------------------------------------------
// WMMA GEMM stage: dst[MT x (16*ntiles)] = act( src[MT x K] @ WT^T + bias )
//   src : LDS, f16, row-major, leading dim ld_src (halves)
//   WT  : global, f16, row-major [N x K] (transposed weights)
// A-fragment per lane (16-bit A 16x32 ISA layout): m = lane%16,
//   lanes 0-15 hold K = {0..7, 16..23}, lanes 16-31 hold K = {8..15, 24..31}
//   -> two contiguous 16B LDS loads per lane.
// B-fragment per lane: n = lane%16, same K order as A
//   -> two contiguous 16B global loads per lane.
// D/C: v8f; element v maps to (m = wave*16 + 8*(lane/16) + v, n = lane%16+16*nt)
// ---------------------------------------------------------------------------
__device__ __forceinline__ void gemm_stage(
    const _Float16* __restrict__ src, int ld_src,
    const _Float16* __restrict__ WT, int K,
    const float* __restrict__ bias, int ntiles,
    _Float16* __restrict__ dst, int ld_dst, bool do_silu) {
  const int lane = threadIdx.x & 31;
  const int wave = threadIdx.x >> 5;
  const int l16  = lane & 15;
  const int hi   = lane >> 4;          // 0 or 1
  const int arow = wave * 16 + l16;    // A-fragment row in the 128-row block
  const int kh   = hi * 8;             // A half-chunk base (halves)
  const int kt_n = K >> 5;             // K tiles of 32

  __builtin_prefetch(WT, 0, 1);        // global_prefetch_b8 on the weight stream

  for (int nt = 0; nt < ntiles; ++nt) {
    const int n = nt * 16 + l16;
    const float bn = bias[n];
    v8f acc = {bn, bn, bn, bn, bn, bn, bn, bn};
    const _Float16* bcol = WT + (size_t)n * K + hi * 8;
    const _Float16* apos = src + arow * ld_src + kh;
    for (int kt = 0; kt < kt_n; ++kt) {
      v8h a0 = *(const v8h*)(apos);
      v8h a1 = *(const v8h*)(apos + 16);
      v16h a = __builtin_shufflevector(a0, a1, 0, 1, 2, 3, 4, 5, 6, 7,
                                       8, 9, 10, 11, 12, 13, 14, 15);
      v8h b0 = *(const v8h*)(bcol + kt * 32);
      v8h b1 = *(const v8h*)(bcol + kt * 32 + 16);
      v16h bf = __builtin_shufflevector(b0, b1, 0, 1, 2, 3, 4, 5, 6, 7,
                                        8, 9, 10, 11, 12, 13, 14, 15);
      acc = __builtin_amdgcn_wmma_f32_16x16x32_f16(
          /*neg_a=*/false, a, /*neg_b=*/false, bf,
          /*c_mod=*/(short)0, acc, /*reuse_a=*/false, /*reuse_b=*/false);
      apos += 32;
    }
    const int mbase = wave * 16 + hi * 8;
#pragma unroll
    for (int v = 0; v < 8; ++v) {
      float x = acc[v];
      if (do_silu) x = silu_f(x);
      dst[(mbase + v) * ld_dst + n] = (_Float16)x;
    }
  }
}

// ---------------------------------------------------------------------------
// Fused main kernel: 800 blocks x 256 threads, 128 rows (b,e) per block.
// Dynamic LDS: bufA/bufB, each 128 x 288 f16 (144 KB total -> 2 WGs / WGP).
// ---------------------------------------------------------------------------
__global__ void k_fused(const float* __restrict__ h_abs,
                        const float* __restrict__ field,
                        const float* __restrict__ Ae,
                        const float* __restrict__ Ab,
                        const float* __restrict__ msq1,
                        const float* __restrict__ msq2,
                        const float* __restrict__ mb1,
                        const _Float16* __restrict__ mW2T,
                        const float* __restrict__ mb2,
                        const _Float16* __restrict__ mW3T,
                        const float* __restrict__ mb3,
                        const _Float16* __restrict__ oW1T,
                        const float* __restrict__ ob1,
                        const _Float16* __restrict__ oW2T,
                        const float* __restrict__ ob2,
                        const float* __restrict__ oW3,
                        const float* __restrict__ ob3,
                        float* __restrict__ out) {
  extern __shared__ _Float16 smem[];
  _Float16* bufA = smem;                 // 128 x 288 halves
  _Float16* bufB = smem + MT * INV_IN;   // 128 x 288 halves

  const int row0 = blockIdx.x * MT;
  const int tid  = threadIdx.x;

  // --- Stage 1: h1 = silu(Ae[e] + Ab[b] + mb1) -> bufA [128 x 256]
  for (int i = tid; i < MT * HIDC; i += blockDim.x) {
    const int r = i / HIDC, n = i % HIDC;
    const int gr = row0 + r;
    const int b = gr / NE_N, e = gr % NE_N;   // row = b*NE + e
    const float x = Ae[e * HIDC + n] + Ab[b * HIDC + n] + mb1[n];
    bufA[r * HIDC + n] = (_Float16)silu_f(x);
  }
  __syncthreads();

  // --- Stage 2: h2 = silu(h1 @ mW2 + mb2) -> bufB [128 x 256]
  gemm_stage(bufA, HIDC, mW2T, HIDC, mb2, HIDC / 16, bufB, HIDC, true);
  __syncthreads();

  // --- Stage 3: gates = h2 @ mW3 + mb3 (linear) -> bufA [128 x 224]
  gemm_stage(bufB, HIDC, mW3T, HIDC, mb3, NCOPC / 16, bufA, NCOPC, false);
  __syncthreads();

  // --- Stage 4: invariant features -> bufB [128 x 288]
  for (int i = tid; i < MT * INV_IN; i += blockDim.x) {
    const int r = i / INV_IN, c = i % INV_IN;
    const int gr = row0 + r;
    const int b = gr / NE_N;
    float v;
    if (c < 128) {                                   // l=0 scalars: x * g
      const float g = (float)bufA[r * NCOPC + c];
      v = h_abs[b * D_TOT + c] * g;
    } else if (c < 192) {                            // l=1: sqrt(g^2*msq + eps)
      const int j = c - 128;
      const float g = (float)bufA[r * NCOPC + 128 + j];
      v = sqrtf(g * g * msq1[b * 64 + j] + EPSC);
    } else if (c < 224) {                            // l=2
      const int j = c - 192;
      const float g = (float)bufA[r * NCOPC + 192 + j];
      v = sqrtf(g * g * msq2[b * 32 + j] + EPSC);
    } else {                                         // field passthrough
      v = field[b * FF_DIMC + (c - 224)];
    }
    bufB[r * INV_IN + c] = (_Float16)v;
  }
  __syncthreads();

  // --- Stage 5: t1 = silu(inv @ oW1 + ob1) -> bufA [128 x 256]
  gemm_stage(bufB, INV_IN, oW1T, INV_IN, ob1, HIDC / 16, bufA, HIDC, true);
  __syncthreads();

  // --- Stage 6: t2 = silu(t1 @ oW2 + ob2) -> bufB [128 x 256]
  gemm_stage(bufA, HIDC, oW2T, HIDC, ob2, HIDC / 16, bufB, HIDC, true);
  __syncthreads();

  // --- Stage 7: out = t2 @ oW3 + ob3 (N=1) : one thread per row
  if (tid < MT) {
    float s = ob3[0];
#pragma unroll 8
    for (int k = 0; k < HIDC; ++k)
      s += (float)bufB[tid * HIDC + k] * oW3[k];
    out[row0 + tid] = s;
  }
}

// ---------------------------------------------------------------------------
// Host-side launch
// ---------------------------------------------------------------------------
extern "C" void kernel_launch(void* const* d_in, const int* in_sizes, int n_in,
                              void* d_out, int out_size, void* d_ws, size_t ws_size,
                              hipStream_t stream) {
  const float* h_abs  = (const float*)d_in[0];   // [256,480]
  const float* e_feat = (const float*)d_in[1];   // [400,32]
  const float* field  = (const float*)d_in[2];   // [256,64]
  const float* mW1    = (const float*)d_in[3];   // [96,256]
  const float* mb1    = (const float*)d_in[4];   // [256]
  const float* mW2    = (const float*)d_in[5];   // [256,256]
  const float* mb2    = (const float*)d_in[6];   // [256]
  const float* mW3    = (const float*)d_in[7];   // [256,224]
  const float* mb3    = (const float*)d_in[8];   // [224]
  const float* oW1    = (const float*)d_in[9];   // [288,256]
  const float* ob1    = (const float*)d_in[10];  // [256]
  const float* oW2    = (const float*)d_in[11];  // [256,256]
  const float* ob2    = (const float*)d_in[12];  // [256]
  const float* oW3    = (const float*)d_in[13];  // [256,1]
  const float* ob3    = (const float*)d_in[14];  // [1]
  float* out = (float*)d_out;

  // Workspace layout (all offsets 64B-aligned)
  char* ws = (char*)d_ws;
  float*    Ae    = (float*)(ws + 0);          // 400*256*4 = 409600
  float*    Ab    = (float*)(ws + 409600);     // 256*256*4 = 262144
  float*    msq1  = (float*)(ws + 671744);     // 256*64*4  = 65536
  float*    msq2  = (float*)(ws + 737280);     // 256*32*4  = 32768
  _Float16* mW2T  = (_Float16*)(ws + 770048);  // 256*256*2 = 131072
  _Float16* mW3T  = (_Float16*)(ws + 901120);  // 224*256*2 = 114688
  _Float16* oW1T  = (_Float16*)(ws + 1015808); // 256*288*2 = 147456
  _Float16* oW2T  = (_Float16*)(ws + 1163264); // 256*256*2 = 131072
                                               // total     = 1294336 B

  const int T = 256;

  // Weight transposes (f32 [K,N] -> f16 [N,K])
  k_transpose_f16<<<(256 * 256 + T - 1) / T, T, 0, stream>>>(mW2, mW2T, 256, 256);
  k_transpose_f16<<<(256 * 224 + T - 1) / T, T, 0, stream>>>(mW3, mW3T, 256, 224);
  k_transpose_f16<<<(288 * 256 + T - 1) / T, T, 0, stream>>>(oW1, oW1T, 288, 256);
  k_transpose_f16<<<(256 * 256 + T - 1) / T, T, 0, stream>>>(oW2, oW2T, 256, 256);

  // Layer-1 factorized partials + per-b mean-square precompute
  k_ae <<<(NE_N * HIDC + T - 1) / T, T, 0, stream>>>(e_feat, mW1, Ae);
  k_ab <<<(B_N * HIDC + T - 1) / T, T, 0, stream>>>(field, mW1, Ab);
  k_msq<<<(B_N * 96 + T - 1) / T, T, 0, stream>>>(h_abs, msq1, msq2);

  // Fused main kernel: 800 blocks x 256 threads, 144KB dynamic LDS
  const size_t shmem = (size_t)2 * MT * INV_IN * sizeof(_Float16); // 147456
  k_fused<<<ROWS / MT, T, shmem, stream>>>(
      h_abs, field, Ae, Ab, msq1, msq2,
      mb1, mW2T, mb2, mW3T, mb3,
      oW1T, ob1, oW2T, ob2, oW3, ob3, out);
}
